// MoEExperts_40407052320887
// MI455X (gfx1250) — matrix operand order repaired and checked by reading, not compiled, not run on hardware
//
#include <hip/hip_runtime.h>
#include <hip/hip_bf16.h>

// MoE experts for MI455X (gfx1250, wave32, WMMA + TDM async tensor loads).
#define NTOK 8192
#define TOPK 2
#define NEXP 8
#define HDIM 2048
#define IDIM 1408
#define I2   (2 * IDIM)
#define NK   (NTOK * TOPK)

typedef __attribute__((ext_vector_type(16))) __bf16 v16bf;
typedef __attribute__((ext_vector_type(8)))  __bf16 v8bf;
typedef __attribute__((ext_vector_type(8)))  float  v8f;
typedef __attribute__((ext_vector_type(4)))  unsigned int u32x4;
typedef __attribute__((ext_vector_type(8)))  int i32x8;
typedef __attribute__((ext_vector_type(4)))  int i32x4;

// LDS tile strides: TK=32 bf16 (64B) plus 16B pad -> 40 elements (80B = 20 banks);
// m*20 mod 64 is a permutation over 16 consecutive rows (conflict-free frags).
#define LDA 40
#define LDB 40
#define ATILE (128 * LDA)   // one A buffer, in bf16 elements

#define WMMA_BF16(a, b, c) \
  __builtin_amdgcn_wmma_f32_16x16x32_bf16(false, (a), false, (b), (short)0, (c), false, false)

__device__ __forceinline__ v8f zero8() {
  v8f z;
#pragma unroll
  for (int i = 0; i < 8; ++i) z[i] = 0.0f;
  return z;
}

// ---- TDM: async 2D bf16 tile load global -> LDS with row padding ----
// Loads tile_rows x 32 bf16 elements; rows beyond valid_rows are zero-filled by
// the TDM OOB rule (tensor_dim1 = valid_rows). LDS rows padded 64B->80B via
// pad_interval=3 (16 DWORDs) / pad_amount=3 (4 DWORDs), matching LDA=40.
__device__ __forceinline__ void tdm_load_a_tile(unsigned lds_off, const __bf16* gptr,
                                                int valid_rows, int row_stride_elems) {
  unsigned long long ga = (unsigned long long)(size_t)gptr;
  u32x4 g0;
  g0[0] = 1u;                                   // count=1, load, no gather
  g0[1] = lds_off;                              // lds_addr (bytes)
  g0[2] = (unsigned)(ga & 0xFFFFFFFFu);         // global_addr[31:0]
  g0[3] = (unsigned)((ga >> 32) & 0x1FFFFFFu) | (2u << 30);  // addr[56:32], type=2
  i32x8 g1;
  g1[0] = (1 << 16) | (1 << 20) | (3 << 22) | (3 << 25); // data_size=2B, pad_en, int=16dw, amt=4dw
  g1[1] = (32 & 0xFFFF) << 16;                  // atomic_bar_addr=0 | tensor_dim0(=32) lo16
  g1[2] = 0 | ((valid_rows & 0xFFFF) << 16);    // tensor_dim0 hi16 | tensor_dim1 lo16
  g1[3] = 0 | (32 << 16);                       // tensor_dim1 hi16 | tile_dim0=32
  g1[4] = 128;                                  // tile_dim1=128 rows | tile_dim2=0
  g1[5] = row_stride_elems;                     // tensor_dim0_stride[31:0]
  g1[6] = 0;                                    // stride hi16 | tensor_dim1_stride lo16 (unused 2D)
  g1[7] = 0;
  i32x4 z4 = {0, 0, 0, 0};
#if defined(__clang_major__) && (__clang_major__ >= 23)
  i32x8 z8 = {0, 0, 0, 0, 0, 0, 0, 0};
  __builtin_amdgcn_tensor_load_to_lds(g0, g1, z4, z4, z8, 0);
#else
  __builtin_amdgcn_tensor_load_to_lds(g0, g1, z4, z4, 0);
#endif
}

// A fragment (16x32 bf16): lanes 0-15 hold M=lane, K=0..7 & 16..23; lanes 16-31
// hold M=lane-16, K=8..15 & 24..31 (pairs K-contiguous -> two 16B LDS reads).
__device__ __forceinline__ v16bf frag_a(const __bf16* base, int lane) {
  int m  = lane & 15;
  int ko = (lane >> 4) << 3;
  const __bf16* p = base + m * LDA;
  v8bf lo = *(const v8bf*)(p + ko);
  v8bf hi = *(const v8bf*)(p + 16 + ko);
  v16bf r;
#pragma unroll
  for (int i = 0; i < 8; ++i) { r[i] = lo[i]; r[i + 8] = hi[i]; }
  return r;
}

// B fragment (32x16 bf16), staged K-contiguous per column: lanes 0-15 hold
// N=lane, K=0..7 & 16..23; lanes 16-31 hold N=lane-16, K=8..15 & 24..31.
__device__ __forceinline__ v16bf frag_b(const __bf16* base, int lane) {
  int n  = lane & 15;
  int kb = (lane >> 4) << 3;
  const __bf16* p = base + n * LDB + kb;
  v8bf lo = *(const v8bf*)(p);
  v8bf hi = *(const v8bf*)(p + 16);
  v16bf r;
#pragma unroll
  for (int i = 0; i < 8; ++i) { r[i] = lo[i]; r[i + 8] = hi[i]; }
  return r;
}

// ---------------- routing kernels ----------------

__global__ void zero_kernel(float* __restrict__ out, int* __restrict__ hdr, long n) {
  long i = (long)blockIdx.x * blockDim.x + threadIdx.x;
  long stride = (long)gridDim.x * blockDim.x;
  for (; i < n; i += stride) out[i] = 0.0f;
  if (blockIdx.x == 0 && threadIdx.x < 32) hdr[threadIdx.x] = 0;
}

__global__ void count_kernel(const int* __restrict__ idx, int* __restrict__ counts) {
  int s = blockIdx.x * blockDim.x + threadIdx.x;
  if (s < NK) atomicAdd(&counts[idx[s]], 1);
}

__global__ void scan_kernel(const int* __restrict__ counts, int* __restrict__ offs) {
  if (threadIdx.x == 0) {
    int acc = 0;
#pragma unroll
    for (int e = 0; e < NEXP; ++e) { offs[e] = acc; acc += counts[e]; }
    offs[NEXP] = acc;
  }
}

__global__ void scatter_kernel(const int* __restrict__ idx, const float* __restrict__ w,
                               const int* __restrict__ offs, int* __restrict__ fill,
                               int* __restrict__ toks, float* __restrict__ wsort) {
  int s = blockIdx.x * blockDim.x + threadIdx.x;
  if (s < NK) {
    int e = idx[s];
    int pos = offs[e] + atomicAdd(&fill[e], 1);
    toks[pos]  = s / TOPK;
    wsort[pos] = w[s];
  }
}

__global__ void gather_kernel(const float* __restrict__ x, const int* __restrict__ toks,
                              __bf16* __restrict__ xs) {
  int p = blockIdx.x;
  int t = toks[p];
  const float* src = x + (size_t)t * HDIM;
  __bf16* dst = xs + (size_t)p * HDIM;
#pragma unroll
  for (int j = 0; j < HDIM / 256; ++j) {
    int i = j * 256 + threadIdx.x;
    dst[i] = (__bf16)src[i];
  }
}

// ---------------- GEMM1: xs @ gate_up -> SwiGLU -> inter (bf16) ----------------
// 128x64 block tile computing BOTH gate and up panels; A tiles via TDM async
// (double-buffered LDS), B panels via fp32 load + cvt + LDS (register prefetch).
__global__ __launch_bounds__(256) void gemm1_swiglu(
    const __bf16* __restrict__ xs, const float* __restrict__ gup,
    __bf16* __restrict__ inter, const int* __restrict__ offs) {
  int e  = blockIdx.z;
  int g0 = offs[e];
  int M  = offs[e + 1] - g0;
  int mt = blockIdx.x;
  if (mt * 128 >= M) return;
  int nt = blockIdx.y;

  __shared__ __bf16 Asm[2 * ATILE];
  __shared__ __bf16 Bg[64 * LDB];
  __shared__ __bf16 Bu[64 * LDB];

  int tid  = threadIdx.x;
  int lane = tid & 31;
  int wv   = tid >> 5;
  int wm   = wv >> 1;
  int wn   = wv & 1;
  bool issuer = (tid < 32);  // wave 0 issues TDM (per-wave op, EXEC ignored)

  int vrows = M - mt * 128; if (vrows > 128) vrows = 128;
  const __bf16* abase = xs + (size_t)(g0 + mt * 128) * HDIM;
  unsigned ldsA0 = (unsigned)(size_t)&Asm[0];
  unsigned ldsA1 = (unsigned)(size_t)&Asm[ATILE];

  // B staging: thread -> (col n, 8 consecutive K); coalesced across lanes per K-row.
  int bn  = tid & 63;
  int bkq = (tid >> 6) << 3;
  const float* wbase = gup + (size_t)e * HDIM * I2;
  const float* gptr  = wbase + (size_t)bkq * I2 + nt * 64 + bn;
  const float* uptr  = gptr + IDIM;

  float gr[8], ur[8];
  auto loadGB = [&](int kk) {
    const float* gp = gptr + (size_t)kk * I2;
    const float* up = uptr + (size_t)kk * I2;
#pragma unroll
    for (int j = 0; j < 8; ++j) {
      gr[j] = gp[(size_t)j * I2];
      ur[j] = up[(size_t)j * I2];
    }
  };
  auto storeSB = [&]() {
    v8bf g8, u8;
#pragma unroll
    for (int j = 0; j < 8; ++j) { g8[j] = (__bf16)gr[j]; u8[j] = (__bf16)ur[j]; }
    *(v8bf*)&Bg[bn * LDB + bkq] = g8;
    *(v8bf*)&Bu[bn * LDB + bkq] = u8;
  };

  v8f accG[2][2], accU[2][2];
#pragma unroll
  for (int r = 0; r < 2; ++r)
#pragma unroll
    for (int c = 0; c < 2; ++c) { accG[r][c] = zero8(); accU[r][c] = zero8(); }

  if (issuer) tdm_load_a_tile(ldsA0, abase, vrows, HDIM);
  loadGB(0);

  int ib = 0;
  for (int kk = 0; kk < HDIM; kk += 32) {
    storeSB();                                   // B regs -> LDS
    if (issuer) __builtin_amdgcn_s_wait_tensorcnt((short)0);  // A buf[ib] landed
    __syncthreads();
    if (kk + 32 < HDIM) {                        // overlap next A DMA + B loads
      if (issuer) tdm_load_a_tile(ib ? ldsA0 : ldsA1, abase + (kk + 32), vrows, HDIM);
      loadGB(kk + 32);
    }

    const __bf16* ab = Asm + ib * ATILE + (wm * 32) * LDA;
    v16bf a0 = frag_a(ab, lane);
    v16bf a1 = frag_a(ab + 16 * LDA, lane);
    const __bf16* gb = Bg + (wn * 32) * LDB;
    v16bf bg0 = frag_b(gb, lane);
    v16bf bg1 = frag_b(gb + 16 * LDB, lane);
    const __bf16* ub = Bu + (wn * 32) * LDB;
    v16bf bu0 = frag_b(ub, lane);
    v16bf bu1 = frag_b(ub + 16 * LDB, lane);

    accG[0][0] = WMMA_BF16(a0, bg0, accG[0][0]);
    accG[0][1] = WMMA_BF16(a0, bg1, accG[0][1]);
    accG[1][0] = WMMA_BF16(a1, bg0, accG[1][0]);
    accG[1][1] = WMMA_BF16(a1, bg1, accG[1][1]);
    accU[0][0] = WMMA_BF16(a0, bu0, accU[0][0]);
    accU[0][1] = WMMA_BF16(a0, bu1, accU[0][1]);
    accU[1][0] = WMMA_BF16(a1, bu0, accU[1][0]);
    accU[1][1] = WMMA_BF16(a1, bu1, accU[1][1]);
    __syncthreads();
    ib ^= 1;
  }

  // Fused SwiGLU epilogue -> bf16 intermediate.
  int nsub = lane & 15;
  int mb   = mt * 128 + wm * 32 + ((lane >> 4) << 3);
#pragma unroll
  for (int r = 0; r < 2; ++r)
#pragma unroll
    for (int c = 0; c < 2; ++c) {
      int col = nt * 64 + wn * 32 + c * 16 + nsub;
#pragma unroll
      for (int i = 0; i < 8; ++i) {
        int m = mb + r * 16 + i;
        if (m < M) {
          float gv = accG[r][c][i];
          float uv = accU[r][c][i];
          float s  = 1.0f / (1.0f + __expf(-gv));
          inter[(size_t)(g0 + m) * IDIM + col] = (__bf16)(gv * s * uv);
        }
      }
    }
}

// ---------------- GEMM2: inter @ down -> scaled scatter-add into out ----------------
__global__ __launch_bounds__(256) void gemm2_kernel(
    const __bf16* __restrict__ inter, const float* __restrict__ down,
    const int* __restrict__ toks, const float* __restrict__ wsort,
    const int* __restrict__ offs, float* __restrict__ out) {
  int e  = blockIdx.z;
  int g0 = offs[e];
  int M  = offs[e + 1] - g0;
  int mt = blockIdx.x;
  if (mt * 128 >= M) return;
  int nt = blockIdx.y;

  __shared__ __bf16 Asm[2 * ATILE];
  __shared__ __bf16 Bsm[64 * LDB];

  int tid  = threadIdx.x;
  int lane = tid & 31;
  int wv   = tid >> 5;
  int wm   = wv >> 1;
  int wn   = wv & 1;
  bool issuer = (tid < 32);

  int vrows = M - mt * 128; if (vrows > 128) vrows = 128;
  const __bf16* abase = inter + (size_t)(g0 + mt * 128) * IDIM;
  unsigned ldsA0 = (unsigned)(size_t)&Asm[0];
  unsigned ldsA1 = (unsigned)(size_t)&Asm[ATILE];

  int bn  = tid & 63;
  int bkq = (tid >> 6) << 3;
  const float* wb   = down + (size_t)e * IDIM * HDIM;
  const float* bptr = wb + (size_t)bkq * HDIM + nt * 64 + bn;

  float br[8];
  auto loadGB = [&](int kk) {
    const float* bp = bptr + (size_t)kk * HDIM;
#pragma unroll
    for (int j = 0; j < 8; ++j) br[j] = bp[(size_t)j * HDIM];
  };
  auto storeSB = [&]() {
    v8bf b8;
#pragma unroll
    for (int j = 0; j < 8; ++j) b8[j] = (__bf16)br[j];
    *(v8bf*)&Bsm[bn * LDB + bkq] = b8;
  };

  v8f acc[2][2];
#pragma unroll
  for (int r = 0; r < 2; ++r)
#pragma unroll
    for (int c = 0; c < 2; ++c) acc[r][c] = zero8();

  if (issuer) tdm_load_a_tile(ldsA0, abase, vrows, IDIM);
  loadGB(0);

  int ib = 0;
  for (int kk = 0; kk < IDIM; kk += 32) {
    storeSB();
    if (issuer) __builtin_amdgcn_s_wait_tensorcnt((short)0);
    __syncthreads();
    if (kk + 32 < IDIM) {
      if (issuer) tdm_load_a_tile(ib ? ldsA0 : ldsA1, abase + (kk + 32), vrows, IDIM);
      loadGB(kk + 32);
    }

    const __bf16* ab = Asm + ib * ATILE + (wm * 32) * LDA;
    v16bf a0 = frag_a(ab, lane);
    v16bf a1 = frag_a(ab + 16 * LDA, lane);
    const __bf16* bb = Bsm + (wn * 32) * LDB;
    v16bf b0 = frag_b(bb, lane);
    v16bf b1 = frag_b(bb + 16 * LDB, lane);

    acc[0][0] = WMMA_BF16(a0, b0, acc[0][0]);
    acc[0][1] = WMMA_BF16(a0, b1, acc[0][1]);
    acc[1][0] = WMMA_BF16(a1, b0, acc[1][0]);
    acc[1][1] = WMMA_BF16(a1, b1, acc[1][1]);
    __syncthreads();
    ib ^= 1;
  }

  int nsub = lane & 15;
  int mb   = mt * 128 + wm * 32 + ((lane >> 4) << 3);
#pragma unroll
  for (int r = 0; r < 2; ++r)
#pragma unroll
    for (int c = 0; c < 2; ++c) {
      int col = nt * 64 + wn * 32 + c * 16 + nsub;
#pragma unroll
      for (int i = 0; i < 8; ++i) {
        int m = mb + r * 16 + i;
        if (m < M) {
          float wgt = wsort[g0 + m];
          int tok   = toks[g0 + m];
          atomicAdd(&out[(size_t)tok * HDIM + col], acc[r][c][i] * wgt);
        }
      }
    }
}

// ---------------- launch ----------------

extern "C" void kernel_launch(void* const* d_in, const int* in_sizes, int n_in,
                              void* d_out, int out_size, void* d_ws, size_t ws_size,
                              hipStream_t stream) {
  const float* x        = (const float*)d_in[0];
  const int*   topk_idx = (const int*)d_in[1];
  const float* topk_w   = (const float*)d_in[2];
  const float* gup      = (const float*)d_in[3];
  const float* down     = (const float*)d_in[4];
  float* out = (float*)d_out;

  // Workspace (~110 MB): header | toks[NK] | wsort[NK] | xs bf16[NK,H] | inter bf16[NK,I]
  int* counts = (int*)d_ws;
  int* fill   = counts + 8;
  int* offs   = counts + 16;
  int* toks   = counts + 64;
  float*  wsort = (float*)(toks + NK);
  __bf16* xs    = (__bf16*)(wsort + NK);
  __bf16* inter = xs + (size_t)NK * HDIM;

  zero_kernel<<<16384, 256, 0, stream>>>(out, counts, (long)NTOK * HDIM);
  count_kernel<<<NK / 256, 256, 0, stream>>>(topk_idx, counts);
  scan_kernel<<<1, 32, 0, stream>>>(counts, offs);
  scatter_kernel<<<NK / 256, 256, 0, stream>>>(topk_idx, topk_w, offs, fill, toks, wsort);
  gather_kernel<<<NK, 256, 0, stream>>>(x, toks, xs);

  dim3 g1(NK / 128, IDIM / 64, NEXP);   // (128, 22, 8); dead tiles early-exit
  gemm1_swiglu<<<g1, 256, 0, stream>>>(xs, gup, inter, offs);

  dim3 g2(NK / 128, HDIM / 64, NEXP);   // (128, 32, 8)
  gemm2_kernel<<<g2, 256, 0, stream>>>(inter, down, toks, wsort, offs, out);
}
